// CausalSelfAttention_11613591568700
// MI455X (gfx1250) — compile-verified, matrix operand order checked by reading, not run on hardware
//
#include <hip/hip_runtime.h>
#include <hip/hip_bf16.h>

#define BB 2
#define TT 2048
#define CC 1024
#define HH 16
#define DD 64
#define BH (BB*HH)
#define MTOT (BB*TT)

typedef __attribute__((ext_vector_type(16))) __bf16 bf16x16;
typedef __attribute__((ext_vector_type(8)))  __bf16 bf16x8;
typedef __attribute__((ext_vector_type(4)))  __bf16 bf16x4;
typedef __attribute__((ext_vector_type(8)))  float  f32x8;
typedef __attribute__((ext_vector_type(4)))  float  f32x4;
typedef int v4i __attribute__((vector_size(16)));   // matches builtin param type

#if __has_builtin(__builtin_amdgcn_global_load_async_to_lds_b128)
#define USE_ASYNC_LDS 1
#endif

static __device__ __forceinline__ __bf16 to_bf16(float f) {
  unsigned u = __builtin_bit_cast(unsigned, f);
  unsigned r = u + 0x7FFFu + ((u >> 16) & 1u);   // round-to-nearest-even
  unsigned short h = (unsigned short)(r >> 16);
  return __builtin_bit_cast(__bf16, h);
}

static __device__ __forceinline__ float elu1(float v) {
  return v > 0.f ? v + 1.f : __expf(v);          // elu(x)+1
}

static __device__ __forceinline__ f32x8 zero8() {
  f32x8 z;
#pragma unroll
  for (int i = 0; i < 8; ++i) z[i] = 0.f;
  return z;
}

// 16-byte global -> LDS copy; async on gfx1250 when the builtin exists.
static __device__ __forceinline__ void copy16(const __bf16* g, __bf16* l) {
#if defined(USE_ASYNC_LDS)
  __builtin_amdgcn_global_load_async_to_lds_b128(
      (__attribute__((address_space(1))) v4i*)(__bf16*)g,
      (__attribute__((address_space(3))) v4i*)l, 0, 0);
#else
  *(bf16x8*)l = *(const bf16x8*)g;
#endif
}

static __device__ __forceinline__ void async_join() {
#if defined(USE_ASYNC_LDS)
#if __has_builtin(__builtin_amdgcn_s_wait_asynccnt)
  __builtin_amdgcn_s_wait_asynccnt(0);
#else
  asm volatile("s_wait_asynccnt 0" ::: "memory");
#endif
#endif
}

// Load a 16x32 bf16 WMMA A/B fragment from row-major [16][stride] storage.
// Wave32 layout: lane L holds row (L&15); lanes 16-31 offset K by 8;
// halfwords 0..7 -> K=kb..kb+7, halfwords 8..15 -> K=kb+16..kb+23.
static __device__ __forceinline__ bf16x16 load_frag(const __bf16* p0, int stride) {
  int lane = threadIdx.x & 31;
  const __bf16* p = p0 + (lane & 15) * stride + ((lane >> 4) << 3);
  bf16x8 lo = *(const bf16x8*)(p);
  bf16x8 hi = *(const bf16x8*)(p + 16);
  bf16x16 v;
#pragma unroll
  for (int i = 0; i < 8; ++i) { v[i] = lo[i]; v[i + 8] = hi[i]; }
  return v;
}

static __device__ __forceinline__ f32x8 wmma_bf16(bf16x16 a, bf16x16 b, f32x8 c) {
  return __builtin_amdgcn_wmma_f32_16x16x32_bf16(
      /*neg_a=*/false, a, /*neg_b=*/false, b,
      /*c_mod=*/(short)0, c, /*reuse_a=*/false, /*reuse_b=*/false);
}

// ---------------------------------------------------------------------------
// Shared double-buffered 128x128 GEMM mainloop, K = CC = 1024.
// Aglob: bf16 [128 rows][CC], row stride CC, pre-offset to (bm, 0).
// Bglob: bf16 [128 rows][CC] ([n][k] layout), pre-offset to (bn, 0).
// ---------------------------------------------------------------------------
static __device__ __forceinline__ void stage_tiles(
    const __bf16* __restrict__ Ag, const __bf16* __restrict__ Bg,
    __bf16* As, __bf16* Bs, int tid)
{
  // two 128x32 bf16 tiles = 2 x 512 16B chunks; 256 threads x 2 chunks each
#pragma unroll
  for (int p = 0; p < 2; ++p) {
    int q = tid + p * 256;
    int r = q >> 2;            // row 0..127
    int c = (q & 3) * 8;       // 0,8,16,24 (elements)
    copy16(Ag + (size_t)r * CC + c, As + r * 32 + c);
    copy16(Bg + (size_t)r * CC + c, Bs + r * 32 + c);
  }
}

static __device__ __forceinline__ void gemm_mainloop_128x128(
    const __bf16* __restrict__ Aglob, const __bf16* __restrict__ Bglob,
    __bf16* As, __bf16* Bs,    // each [2][128*32]
    f32x8 acc[4][2])
{
  int tid = threadIdx.x;
  int wave = tid >> 5, wm = wave >> 2, wn = wave & 3;   // 2 x 4 wave grid

  stage_tiles(Aglob, Bglob, As, Bs, tid);
  async_join();
  __syncthreads();

  for (int kk = 0; kk < CC; kk += 32) {
    int cur = (kk >> 5) & 1;
    __bf16* Ac = As + cur * (128 * 32);
    __bf16* Bc = Bs + cur * (128 * 32);
    if (kk + 32 < CC) {
      int nxt = cur ^ 1;
      stage_tiles(Aglob + kk + 32, Bglob + kk + 32,
                  As + nxt * (128 * 32), Bs + nxt * (128 * 32), tid);
    }
    bf16x16 af[4], bfr[2];
#pragma unroll
    for (int mi = 0; mi < 4; ++mi) af[mi]  = load_frag(Ac + (wm * 64 + mi * 16) * 32, 32);
#pragma unroll
    for (int ni = 0; ni < 2; ++ni) bfr[ni] = load_frag(Bc + (wn * 32 + ni * 16) * 32, 32);
#pragma unroll
    for (int mi = 0; mi < 4; ++mi)
#pragma unroll
      for (int ni = 0; ni < 2; ++ni)
        acc[mi][ni] = wmma_bf16(af[mi], bfr[ni], acc[mi][ni]);
    async_join();
    __syncthreads();
  }
}

// ---------------------------------------------------------------------------
// Prepass: f32 -> bf16 elementwise (vectorized x4)
// ---------------------------------------------------------------------------
__global__ __launch_bounds__(256) void k_cvt(
    const float* __restrict__ src, __bf16* __restrict__ dst, int n4)
{
  int i = blockIdx.x * 256 + threadIdx.x;
  if (i >= n4) return;
  f32x4 v = *(const f32x4*)(src + (size_t)i * 4);
  bf16x4 o;
#pragma unroll
  for (int j = 0; j < 4; ++j) o[j] = to_bf16(v[j]);
  *(bf16x4*)(dst + (size_t)i * 4) = o;
}

// ---------------------------------------------------------------------------
// Prepass: f32 [K][N] -> bf16 [N][K] tile transpose (64x64 tiles via LDS)
// ---------------------------------------------------------------------------
__global__ __launch_bounds__(256) void k_transpose(
    const float* __restrict__ src, __bf16* __restrict__ dst, int K, int N)
{
  __shared__ __bf16 Ts[64][65];
  int k0 = blockIdx.x * 64, n0 = blockIdx.y * 64;
  int tid = threadIdx.x;
#pragma unroll
  for (int p = 0; p < 16; ++p) {
    int idx = tid + p * 256;
    int lk = idx >> 6, ln = idx & 63;
    Ts[ln][lk] = to_bf16(src[(size_t)(k0 + lk) * N + n0 + ln]);   // coalesced over n
  }
  __syncthreads();
#pragma unroll
  for (int p = 0; p < 16; ++p) {
    int idx = tid + p * 256;
    int lr = idx >> 6, lc = idx & 63;
    dst[(size_t)(n0 + lr) * K + k0 + lc] = Ts[lr][lc];            // coalesced over k
  }
}

// ---------------------------------------------------------------------------
// Kernel 1: qkv = Xb @ WaT^T + b_attn ; split Q/K/V, elu+1 on Q/K,
//           store Qe,Ke as f32 [bh][t][d], V transposed bf16 [bh][d][t].
// ---------------------------------------------------------------------------
__global__ __launch_bounds__(256) void k_qkv(
    const __bf16* __restrict__ Xb, const __bf16* __restrict__ WaT,
    const float* __restrict__ bias,
    float* __restrict__ Qe, float* __restrict__ Ke, __bf16* __restrict__ VT)
{
  __shared__ __align__(16) __bf16 As[2 * 128 * 32];
  __shared__ __align__(16) __bf16 Bs[2 * 128 * 32];
  int tid = threadIdx.x, lane = tid & 31, wave = tid >> 5;
  int wm = wave >> 2, wn = wave & 3;
  int bm = blockIdx.x * 128, bn = blockIdx.y * 128;

  f32x8 acc[4][2];
#pragma unroll
  for (int a = 0; a < 4; ++a)
#pragma unroll
    for (int b = 0; b < 2; ++b) acc[a][b] = zero8();

  gemm_mainloop_128x128(Xb + (size_t)bm * CC, WaT + (size_t)bn * CC, As, Bs, acc);

  int mlane = 8 * (lane >> 4);
  int nlane = lane & 15;
#pragma unroll
  for (int mi = 0; mi < 4; ++mi)
#pragma unroll
    for (int ni = 0; ni < 2; ++ni)
#pragma unroll
      for (int e = 0; e < 8; ++e) {
        int m = bm + wm * 64 + mi * 16 + e + mlane;
        int n = bn + wn * 32 + ni * 16 + nlane;
        float v = acc[mi][ni][e] + bias[n];
        int b = m >> 11, t = m & (TT - 1);
        int region = n >> 10, c = n & (CC - 1);
        int h = c >> 6, d = c & 63;
        size_t bh = (size_t)(b * HH + h);
        if (region == 0)      Qe[(bh * TT + t) * DD + d] = elu1(v);
        else if (region == 1) Ke[(bh * TT + t) * DD + d] = elu1(v);
        else                  VT[(bh * DD + d) * TT + t] = to_bf16(v);
      }
}

// ---------------------------------------------------------------------------
// Kernel 2: RoPE.  Qr bf16 [bh][t][d], KrT bf16 [bh][d][t].
// ---------------------------------------------------------------------------
__global__ __launch_bounds__(256) void k_rope(
    const float* __restrict__ Qe, const float* __restrict__ Ke,
    __bf16* __restrict__ Qr, __bf16* __restrict__ KrT)
{
  int idx = blockIdx.x * 256 + threadIdx.x;   // BH*TT*DD threads
  int d  = idx & 63;
  int t  = (idx >> 6) & (TT - 1);
  int bh = idx >> 17;
  // inv_freq = 10000^(-(2*(d%32))/64)
  float ang = (float)t * __expf(-(float)(2 * (d & 31)) * (9.210340371976184f / 64.f));
  float cs = cosf(ang), sn = sinf(ang);
  size_t base = ((size_t)bh * TT + t) * DD;
  float sgn = (d < 32) ? -1.f : 1.f;
  float q = Qe[base + d], qp = Qe[base + (d ^ 32)];
  float k = Ke[base + d], kp = Ke[base + (d ^ 32)];
  Qr[base + d] = to_bf16(q * cs + sgn * qp * sn);
  KrT[((size_t)bh * DD + d) * TT + t] = to_bf16(k * cs + sgn * kp * sn);
}

// ---------------------------------------------------------------------------
// Kernel 3: Ksum[bh][d] = sum_t Ke[bh][t][d]
// ---------------------------------------------------------------------------
__global__ __launch_bounds__(256) void k_ksum(
    const float* __restrict__ Ke, float* __restrict__ Ksum)
{
  int idx = blockIdx.x * 256 + threadIdx.x;   // BH*DD = 2048
  if (idx >= BH * DD) return;
  int d = idx & 63, bh = idx >> 6;
  const float* p = Ke + ((size_t)bh * TT) * DD + d;
  float s = 0.f;
  for (int t = 0; t < TT; ++t) s += p[(size_t)t * DD];
  Ksum[idx] = s;
}

// ---------------------------------------------------------------------------
// Kernel 4: denom[bh][t] = Qe[bh][t][:] . Ksum[bh][:]
// ---------------------------------------------------------------------------
__global__ __launch_bounds__(256) void k_denom(
    const float* __restrict__ Qe, const float* __restrict__ Ksum,
    float* __restrict__ denom)
{
  int idx = blockIdx.x * 256 + threadIdx.x;   // BH*TT = 65536
  int t = idx & (TT - 1), bh = idx >> 11;
  const float* q  = Qe + ((size_t)bh * TT + t) * DD;
  const float* ks = Ksum + bh * DD;
  float s = 0.f;
#pragma unroll 8
  for (int d = 0; d < DD; ++d) s += q[d] * ks[d];
  denom[idx] = s;
}

// ---------------------------------------------------------------------------
// Kernel 5: KV = Kr^T V per (b,h): 64x64, K = T = 2048.  A = KrT [d][t],
//           B = VT [e][t] ([n][k]).  Store transposed: KVT[bh][e][d] (bf16).
// ---------------------------------------------------------------------------
__global__ __launch_bounds__(128) void k_kv(
    const __bf16* __restrict__ KrT, const __bf16* __restrict__ VT,
    __bf16* __restrict__ KVT)
{
  int bh = blockIdx.x;
  int lane = threadIdx.x & 31, wave = threadIdx.x >> 5;   // 4 waves, 16 d-rows each
  const __bf16* A  = KrT + (size_t)bh * DD * TT;
  const __bf16* Bm = VT  + (size_t)bh * DD * TT;
  f32x8 acc[4];
#pragma unroll
  for (int i = 0; i < 4; ++i) acc[i] = zero8();

  for (int kt = 0; kt < TT; kt += 32) {
    bf16x16 af = load_frag(A + (size_t)(wave * 16) * TT + kt, TT);
#pragma unroll
    for (int ni = 0; ni < 4; ++ni) {
      bf16x16 bfr = load_frag(Bm + (size_t)(ni * 16) * TT + kt, TT);
      acc[ni] = wmma_bf16(af, bfr, acc[ni]);
    }
  }
#pragma unroll
  for (int ni = 0; ni < 4; ++ni)
#pragma unroll
    for (int e8 = 0; e8 < 8; ++e8) {
      int dI = wave * 16 + e8 + 8 * (lane >> 4);   // M
      int eI = ni * 16 + (lane & 15);              // N
      KVT[((size_t)bh * DD + eI) * DD + dI] = to_bf16(acc[ni][e8]);
    }
}

// ---------------------------------------------------------------------------
// Kernel 6: y = (Qr @ KV) / denom, repacked bf16 [b*T+t][h*64+e]
// ---------------------------------------------------------------------------
__global__ __launch_bounds__(256) void k_y(
    const __bf16* __restrict__ Qr, const __bf16* __restrict__ KVT,
    const float* __restrict__ denom, __bf16* __restrict__ Yb)
{
  int bh = blockIdx.y;
  int bm = blockIdx.x * 128;
  int lane = threadIdx.x & 31, wave = threadIdx.x >> 5;   // 8 waves, 16 t-rows each
  const __bf16* A  = Qr  + ((size_t)bh * TT + bm + wave * 16) * DD;  // [t][d]
  const __bf16* Bm = KVT + (size_t)bh * DD * DD;                     // [e][d]
  f32x8 acc[4];
#pragma unroll
  for (int i = 0; i < 4; ++i) acc[i] = zero8();

#pragma unroll
  for (int kd = 0; kd < DD; kd += 32) {
    bf16x16 af = load_frag(A + kd, DD);
#pragma unroll
    for (int ni = 0; ni < 4; ++ni) {
      bf16x16 bfr = load_frag(Bm + (ni * 16) * DD + kd, DD);
      acc[ni] = wmma_bf16(af, bfr, acc[ni]);
    }
  }
  int b = bh >> 4, h = bh & 15;
#pragma unroll
  for (int ni = 0; ni < 4; ++ni)
#pragma unroll
    for (int e8 = 0; e8 < 8; ++e8) {
      int t = bm + wave * 16 + e8 + 8 * (lane >> 4);
      int eI = ni * 16 + (lane & 15);
      float v = acc[ni][e8] / denom[bh * TT + t];
      Yb[((size_t)(b * TT + t)) * CC + h * DD + eI] = to_bf16(v);
    }
}

// ---------------------------------------------------------------------------
// Kernel 7: out = Yb @ WpT^T + b_proj  (f32 out)
// ---------------------------------------------------------------------------
__global__ __launch_bounds__(256) void k_proj(
    const __bf16* __restrict__ Yb, const __bf16* __restrict__ WpT,
    const float* __restrict__ bias, float* __restrict__ out)
{
  __shared__ __align__(16) __bf16 As[2 * 128 * 32];
  __shared__ __align__(16) __bf16 Bs[2 * 128 * 32];
  int tid = threadIdx.x, lane = tid & 31, wave = tid >> 5;
  int wm = wave >> 2, wn = wave & 3;
  int bm = blockIdx.x * 128, bn = blockIdx.y * 128;

  f32x8 acc[4][2];
#pragma unroll
  for (int a = 0; a < 4; ++a)
#pragma unroll
    for (int b = 0; b < 2; ++b) acc[a][b] = zero8();

  gemm_mainloop_128x128(Yb + (size_t)bm * CC, WpT + (size_t)bn * CC, As, Bs, acc);

  int mlane = 8 * (lane >> 4);
  int nlane = lane & 15;
#pragma unroll
  for (int mi = 0; mi < 4; ++mi)
#pragma unroll
    for (int ni = 0; ni < 2; ++ni)
#pragma unroll
      for (int e = 0; e < 8; ++e) {
        int m = bm + wm * 64 + mi * 16 + e + mlane;
        int n = bn + wn * 32 + ni * 16 + nlane;
        out[(size_t)m * CC + n] = acc[mi][ni][e] + bias[n];
      }
}

// ---------------------------------------------------------------------------
extern "C" void kernel_launch(void* const* d_in, const int* in_sizes, int n_in,
                              void* d_out, int out_size, void* d_ws, size_t ws_size,
                              hipStream_t stream) {
  (void)in_sizes; (void)n_in; (void)out_size; (void)ws_size;
  const float* x      = (const float*)d_in[0];
  const float* w_attn = (const float*)d_in[1];
  const float* b_attn = (const float*)d_in[2];
  const float* w_proj = (const float*)d_in[3];
  const float* b_proj = (const float*)d_in[4];
  float* out = (float*)d_out;

  char* ws = (char*)d_ws;
  size_t off = 0;
  auto carve = [&](size_t bytes) -> void* {
    void* p = ws + off;
    off += (bytes + 255) & ~(size_t)255;
    return p;
  };
  __bf16* Xb    = (__bf16*)carve((size_t)MTOT * CC * 2);        //  8 MB
  __bf16* WaT   = (__bf16*)carve((size_t)3 * CC * CC * 2);      //  6 MB
  __bf16* WpT   = (__bf16*)carve((size_t)CC * CC * 2);          //  2 MB
  float*  Qe    = (float*) carve((size_t)BH * TT * DD * 4);     // 16 MB
  float*  Ke    = (float*) carve((size_t)BH * TT * DD * 4);     // 16 MB
  __bf16* VT    = (__bf16*)carve((size_t)BH * TT * DD * 2);     //  8 MB
  __bf16* Qr    = (__bf16*)carve((size_t)BH * TT * DD * 2);     //  8 MB
  __bf16* KrT   = (__bf16*)carve((size_t)BH * TT * DD * 2);     //  8 MB
  float*  Ksum  = (float*) carve((size_t)BH * DD * 4);
  float*  denom = (float*) carve((size_t)BH * TT * 4);
  __bf16* KVT   = (__bf16*)carve((size_t)BH * DD * DD * 2);
  __bf16* Yb    = (__bf16*)carve((size_t)MTOT * CC * 2);        //  8 MB

  // Prepass: bf16 conversion + weight transposes
  k_cvt      <<<(MTOT * CC / 4) / 256, 256, 0, stream>>>(x, Xb, MTOT * CC / 4);
  k_transpose<<<dim3(CC / 64, 3 * CC / 64), 256, 0, stream>>>(w_attn, WaT, CC, 3 * CC);
  k_transpose<<<dim3(CC / 64, CC / 64),     256, 0, stream>>>(w_proj, WpT, CC, CC);

  // Attention pipeline
  k_qkv  <<<dim3(MTOT / 128, (3 * CC) / 128), 256, 0, stream>>>(Xb, WaT, b_attn, Qe, Ke, VT);
  k_rope <<<(BH * TT * DD) / 256, 256, 0, stream>>>(Qe, Ke, Qr, KrT);
  k_ksum <<<(BH * DD + 255) / 256, 256, 0, stream>>>(Ke, Ksum);
  k_denom<<<(BH * TT) / 256, 256, 0, stream>>>(Qe, Ksum, denom);
  k_kv   <<<BH, 128, 0, stream>>>(KrT, VT, KVT);
  k_y    <<<dim3(TT / 128, BH), 256, 0, stream>>>(Qr, KVT, denom, Yb);
  k_proj <<<dim3(MTOT / 128, CC / 128), 256, 0, stream>>>(Yb, WpT, b_proj, out);
}
